// SOTTransform_77996606095480
// MI455X (gfx1250) — matrix operand order repaired and checked by reading, not compiled, not run on hardware
//
#include <hip/hip_runtime.h>
#include <hip/hip_bf16.h>

// ---------------------------------------------------------------------------
// SOT transform for MI455X (gfx1250, wave32).
//   K = exp(-LAMB*(2 - 2*(x/|x|)(x/|x|)^T + ALPHA*I))   (8192x8192 fp32, in d_out)
//   100 Sinkhorn iterations of symmetric matvecs (memory-bound, ~2.3 ms roofline)
//   res = u*K*v + I  (in place)
// GEMM uses v_wmma_f32_16x16x32_f16 with LDS-staged tiles.
// ---------------------------------------------------------------------------

#define N_PTS 8192
#define K_DIM 1024
#define LAMB_F 0.1f
#define ALPHA_F 1000000.0f
#define SCALE_F 1024.0f
#define INV_S2 (1.0f / (SCALE_F * SCALE_F))

typedef __attribute__((ext_vector_type(16))) _Float16 v16h;
typedef __attribute__((ext_vector_type(8)))  _Float16 v8h;
typedef __attribute__((ext_vector_type(8)))  float    v8f;
typedef __attribute__((ext_vector_type(4)))  float    vf4;

// ---- init: u=v=1, normsq=0 -------------------------------------------------
__global__ void init_kernel(float* __restrict__ u, float* __restrict__ v,
                            float* __restrict__ normsq) {
    int i = blockIdx.x * blockDim.x + threadIdx.x;
    if (i < N_PTS) { u[i] = 1.0f; v[i] = 1.0f; }
    if (i == 0) *normsq = 0.0f;
}

// ---- sum of squares (Frobenius norm^2) ------------------------------------
__global__ __launch_bounds__(256)
void sumsq_kernel(const float* __restrict__ x, float* __restrict__ normsq, int n) {
    __shared__ float red[256];
    float s = 0.0f;
    for (int i = blockIdx.x * blockDim.x + threadIdx.x; i < n;
         i += gridDim.x * blockDim.x) {
        float t = x[i];
        s += t * t;
    }
    red[threadIdx.x] = s;
    __syncthreads();
    for (int off = 128; off > 0; off >>= 1) {
        if ((int)threadIdx.x < off) red[threadIdx.x] += red[threadIdx.x + off];
        __syncthreads();
    }
    if (threadIdx.x == 0) atomicAdd(normsq, red[0]);
}

// ---- convert to scaled f16: Y = x * (SCALE / |x|_F) -----------------------
__global__ __launch_bounds__(256)
void tohalf_kernel(const float* __restrict__ x, const float* __restrict__ normsq,
                   _Float16* __restrict__ y, int n) {
    const float inv = SCALE_F * rsqrtf(*normsq);
    for (int i = blockIdx.x * blockDim.x + threadIdx.x; i < n;
         i += gridDim.x * blockDim.x) {
        y[i] = (_Float16)(x[i] * inv);
    }
}

// ---- WMMA GEMM + exp epilogue:  Kout = exp(-LAMB*(2 - 2*YY^T/S^2 + A*I)) --
__global__ __launch_bounds__(256)
void gemm_exp_kernel(const _Float16* __restrict__ Y, float* __restrict__ Kout) {
    __shared__ __align__(32) _Float16 ldsA[128 * 32];   // 8 KB: row block panel
    __shared__ __align__(32) _Float16 ldsB[128 * 32];   // 8 KB: col block panel

    const int tid   = threadIdx.x;
    const int wave  = tid >> 5;
    const int lane  = tid & 31;
    const int laneM = lane & 15;   // row/col within 16x16 tile
    const int laneH = lane >> 4;   // half-wave selector

    const int rBase = blockIdx.y * 128;
    const int cBase = blockIdx.x * 128;
    // 8 waves arranged 4x2: each wave owns a 32x64 sub-tile
    const int rOff = (wave >> 1) * 32;
    const int cOff = (wave & 1) * 64;

    v8f acc[2][4];
#pragma unroll
    for (int a = 0; a < 2; ++a)
#pragma unroll
        for (int b = 0; b < 4; ++b) acc[a][b] = (v8f){0,0,0,0,0,0,0,0};

    for (int kk = 0; kk < K_DIM; kk += 32) {
        __syncthreads();
        // Cooperative load: 128 rows x 32 halves (64B) per panel = 512 x 16B chunks.
#pragma unroll
        for (int i = 0; i < 2; ++i) {
            const int chunk = tid * 2 + i;
            const int row   = chunk >> 2;
            const int sub   = chunk & 3;
            const vf4 va = *(const vf4*)(Y + (size_t)(rBase + row) * K_DIM + kk + sub * 8);
            *(vf4*)(ldsA + row * 32 + sub * 8) = va;
            const vf4 vb = *(const vf4*)(Y + (size_t)(cBase + row) * K_DIM + kk + sub * 8);
            *(vf4*)(ldsB + row * 32 + sub * 8) = vb;
        }
        __syncthreads();

        // A fragments (16x32 f16): lane holds row laneM, 16 contiguous K at laneH*16
        v16h afrag[2];
#pragma unroll
        for (int a = 0; a < 2; ++a) {
            const int m = rOff + a * 16 + laneM;
            union { v16h v; v8h h[2]; } u_;
            u_.h[0] = *(const v8h*)(ldsA + m * 32 + laneH * 16);
            u_.h[1] = *(const v8h*)(ldsA + m * 32 + laneH * 16 + 8);
            afrag[a] = u_.v;
        }
        // B fragments (32x16 f16): lane holds col laneM, 16 contiguous K at laneH*16
        v16h bfrag[4];
#pragma unroll
        for (int b = 0; b < 4; ++b) {
            const int n = cOff + b * 16 + laneM;
            bfrag[b] = *(const v16h*)(ldsB + n * 32 + laneH * 16);
        }
#pragma unroll
        for (int a = 0; a < 2; ++a)
#pragma unroll
            for (int b = 0; b < 4; ++b)
                acc[a][b] = __builtin_amdgcn_wmma_f32_16x16x32_f16(
                    false, afrag[a], false, bfrag[b],
                    (short)0, acc[a][b], false, false);
    }

    // Epilogue: C/D layout -> row = j + laneH*8, col = laneM within 16x16 tile.
#pragma unroll
    for (int a = 0; a < 2; ++a) {
#pragma unroll
        for (int b = 0; b < 4; ++b) {
#pragma unroll
            for (int j = 0; j < 8; ++j) {
                const int gr = rBase + rOff + a * 16 + laneH * 8 + j;
                const int gc = cBase + cOff + b * 16 + laneM;
                const float g   = acc[a][b][j] * INV_S2;
                const float arg = -LAMB_F * (2.0f - 2.0f * g +
                                             ((gr == gc) ? ALPHA_F : 0.0f));
                Kout[(size_t)gr * N_PTS + gc] = __expf(arg);
            }
        }
    }
}

// ---- symmetric matvec + reciprocal:  out[r] = 1/(scale[r] * sum_c K[r][c]*x[c])
__global__ __launch_bounds__(256)
void matvec_recip_kernel(const float* __restrict__ Kmat,
                         const float* __restrict__ x,
                         const float* __restrict__ scale,
                         float* __restrict__ out) {
    const int r = blockIdx.x;
    const vf4* __restrict__ Krow = (const vf4*)(Kmat + (size_t)r * N_PTS);
    const vf4* __restrict__ x4   = (const vf4*)x;
    float s = 0.0f;
#pragma unroll 4
    for (int j = threadIdx.x; j < N_PTS / 4; j += 256) {
        __builtin_prefetch(&Krow[j + 512], 0, 1);  // global_prefetch_b8, 2 iters ahead
        const vf4 kv = Krow[j];
        const vf4 xv = x4[j];
        s += kv.x * xv.x + kv.y * xv.y + kv.z * xv.z + kv.w * xv.w;
    }
    __shared__ float red[256];
    red[threadIdx.x] = s;
    __syncthreads();
    for (int off = 128; off > 0; off >>= 1) {
        if ((int)threadIdx.x < off) red[threadIdx.x] += red[threadIdx.x + off];
        __syncthreads();
    }
    if (threadIdx.x == 0) out[r] = 1.0f / (scale[r] * red[0]);
}

// ---- finalize in place:  K <- u[r]*K*v[c] + I ------------------------------
__global__ __launch_bounds__(256)
void finalize_kernel(float* __restrict__ Kmat, const float* __restrict__ u,
                     const float* __restrict__ v) {
    const size_t total4 = (size_t)N_PTS * N_PTS / 4;
    for (size_t i = (size_t)blockIdx.x * blockDim.x + threadIdx.x; i < total4;
         i += (size_t)gridDim.x * blockDim.x) {
        const size_t r  = i >> 11;            // (i*4) / 8192
        const size_t c0 = (i & 2047) << 2;    // column of component .x
        vf4 kv = ((vf4*)Kmat)[i];
        const vf4 vv = ((const vf4*)v)[i & 2047];
        const float ur = u[r];
        kv.x = ur * kv.x * vv.x;
        kv.y = ur * kv.y * vv.y;
        kv.z = ur * kv.z * vv.z;
        kv.w = ur * kv.w * vv.w;
        if (r == c0)     kv.x += 1.0f;
        if (r == c0 + 1) kv.y += 1.0f;
        if (r == c0 + 2) kv.z += 1.0f;
        if (r == c0 + 3) kv.w += 1.0f;
        ((vf4*)Kmat)[i] = kv;
    }
}

// ---------------------------------------------------------------------------
extern "C" void kernel_launch(void* const* d_in, const int* in_sizes, int n_in,
                              void* d_out, int out_size, void* d_ws, size_t ws_size,
                              hipStream_t stream) {
    (void)in_sizes; (void)n_in; (void)out_size; (void)ws_size;

    const float* X   = (const float*)d_in[0];        // 8192 x 1024 fp32
    float* Kmat      = (float*)d_out;                // 8192 x 8192 fp32 (K, then res)

    char* ws         = (char*)d_ws;
    _Float16* Y      = (_Float16*)ws;                            // 16 MB scaled f16
    float* u         = (float*)(ws + (size_t)N_PTS * K_DIM * 2); // 32 KB
    float* vv        = u + N_PTS;                                 // 32 KB
    float* normsq    = vv + N_PTS;                                // 4 B

    const int nElem = N_PTS * K_DIM;

    init_kernel<<<(N_PTS + 255) / 256, 256, 0, stream>>>(u, vv, normsq);
    sumsq_kernel<<<1024, 256, 0, stream>>>(X, normsq, nElem);
    tohalf_kernel<<<2048, 256, 0, stream>>>(X, normsq, Y, nElem);

    dim3 grid(N_PTS / 128, N_PTS / 128);             // 64 x 64 blocks
    gemm_exp_kernel<<<grid, 256, 0, stream>>>(Y, Kmat);

    // Sinkhorn: K symmetric, so K^T@u == K@u. u <- 1/(u*(Kv)); v <- 1/(v*(Ku)).
    for (int it = 0; it < 100; ++it) {
        matvec_recip_kernel<<<N_PTS, 256, 0, stream>>>(Kmat, vv, u, u);
        matvec_recip_kernel<<<N_PTS, 256, 0, stream>>>(Kmat, u, vv, vv);
    }

    finalize_kernel<<<16384, 256, 0, stream>>>(Kmat, u, vv);
}
